// GuidedAttention_38345468019364
// MI455X (gfx1250) — compile-verified, matrix operand order checked by reading, not run on hardware
//
#include <hip/hip_runtime.h>
#include <hip/hip_bf16.h>

// ---------------------------------------------------------------------------
// GuidedAttention for MI455X (gfx1250):
//   bf16 WMMA everywhere, TDM double-buffered LDS staging, flash-softmax.
// ---------------------------------------------------------------------------

#define B_   4
#define TA_  1024
#define TV_  1024
#define D_   512
#define H_   8
#define HD_  64
#define SCALE_ 0.125f   // HD^-0.5
#define LN_EPS_ 1e-5f

typedef __attribute__((ext_vector_type(16))) __bf16 v16bf;
typedef __attribute__((ext_vector_type(8)))  __bf16 bf16x8;
typedef __attribute__((ext_vector_type(8)))  float  v8f;
typedef __attribute__((ext_vector_type(4)))  unsigned int u32x4;
typedef __attribute__((ext_vector_type(8)))  int i32x8;
typedef __attribute__((ext_vector_type(4)))  int i32x4;

#if __has_builtin(__builtin_amdgcn_tensor_load_to_lds)
#define HAVE_TDM 1
#else
#define HAVE_TDM 0
#endif

// ---------------------------------------------------------------------------
// Operand fragment loaders. CDNA5 16-bit A/B operand layout (wave32):
//   lane half selects K sub-block (+8), VGPR i packs (K,K+1) with
//   K = kb + 2i + (i>=4 ? 8 : 0)  ->  elems [kb..kb+7] and [kb+16..kb+23].
// ---------------------------------------------------------------------------
__device__ __forceinline__ v16bf frag_from_f32(const float* rowp, int k0, int lane) {
  const int kb = k0 + ((lane & 16) ? 8 : 0);
  const float4 q0 = *reinterpret_cast<const float4*>(rowp + kb + 0);
  const float4 q1 = *reinterpret_cast<const float4*>(rowp + kb + 4);
  const float4 q2 = *reinterpret_cast<const float4*>(rowp + kb + 16);
  const float4 q3 = *reinterpret_cast<const float4*>(rowp + kb + 20);
  const float f[16] = {q0.x, q0.y, q0.z, q0.w, q1.x, q1.y, q1.z, q1.w,
                       q2.x, q2.y, q2.z, q2.w, q3.x, q3.y, q3.z, q3.w};
  v16bf a;
#pragma unroll
  for (int i = 0; i < 16; ++i) a[i] = (__bf16)f[i];
  return a;
}

__device__ __forceinline__ v16bf frag_from_bf16(const __bf16* rowp, int k0, int lane) {
  const int kb = k0 + ((lane & 16) ? 8 : 0);
  const bf16x8 p0 = *reinterpret_cast<const bf16x8*>(rowp + kb);
  const bf16x8 p1 = *reinterpret_cast<const bf16x8*>(rowp + kb + 16);
  v16bf a;
#pragma unroll
  for (int i = 0; i < 8; ++i) { a[i] = p0[i]; a[i + 8] = p1[i]; }
  return a;
}

#define WMMA_BF16(A, Bm, C) \
  __builtin_amdgcn_wmma_f32_16x16x32_bf16(false, (A), false, (Bm), (short)0, (C), false, false)

// ---------------------------------------------------------------------------
// TDM: 2D tile DMA global -> LDS (Tensor DMA Descriptor per CDNA5 ISA §8).
// Sizes/strides in data_size units; dsz_code: 0=1B, 1=2B, 2=4B.
// ---------------------------------------------------------------------------
#if HAVE_TDM
__device__ __forceinline__ void tdm_load_2d(unsigned lds_off, const void* gptr,
                                            unsigned dsz_code,
                                            unsigned tile_d0, unsigned tile_d1,
                                            unsigned tensor_d0, unsigned tensor_d1,
                                            unsigned stride_d0) {
  const unsigned long long ga = (unsigned long long)gptr;
  u32x4 g0;
  g0[0] = 1u;                                               // count=1 (valid user D#)
  g0[1] = lds_off;                                          // lds_addr
  g0[2] = (unsigned)(ga & 0xffffffffu);                     // global_addr[31:0]
  g0[3] = (unsigned)((ga >> 32) & 0x1ffffffu) | (2u << 30); // ga[56:32] | type=2
  i32x8 g1;
  g1[0] = (int)(dsz_code << 16);                            // data_size, mask=0
  g1[1] = (int)((tensor_d0 & 0xffffu) << 16);               // tensor_dim0[15:0]
  g1[2] = (int)((tensor_d0 >> 16) | ((tensor_d1 & 0xffffu) << 16));
  g1[3] = (int)((tensor_d1 >> 16) | ((tile_d0 & 0xffffu) << 16));
  g1[4] = (int)(tile_d1 & 0xffffu);                         // tile_dim1, tile_dim2=0
  g1[5] = (int)stride_d0;                                   // tensor_dim0_stride[31:0]
  g1[6] = 0;
  g1[7] = 0;
  const i32x4 gz = {0, 0, 0, 0};
#if defined(__clang_major__) && (__clang_major__ >= 23)
  const i32x8 gz8 = {0, 0, 0, 0, 0, 0, 0, 0};
  __builtin_amdgcn_tensor_load_to_lds(g0, g1, gz, gz, gz8, 0);
#else
  __builtin_amdgcn_tensor_load_to_lds(g0, g1, gz, gz, 0);
#endif
}
#endif

__device__ __forceinline__ void tdm_wait() {
#if __has_builtin(__builtin_amdgcn_s_wait_tensorcnt)
  __builtin_amdgcn_s_wait_tensorcnt(0);
#else
  asm volatile("s_wait_tensorcnt 0x0" ::: "memory");
#endif
}

// ---------------------------------------------------------------------------
// Kernel 1: Y = X @ W^T + bias, f32 in -> bf16 out.
//   mode 0: Y laid out [B,H,T,HD]   (Q, K)
//   mode 1: Y laid out [B,H,HD,TV]  (V transposed for P@V operand loads)
// Block = 128 threads (4 waves) computes a 64x64 tile.
// Per 32-wide K step, TDM double-buffers the 64x32 X and W tiles in LDS:
// wave 0 DMAs buffer cur^1 while all waves compute from buffer cur.
// ---------------------------------------------------------------------------
__global__ __launch_bounds__(128) void proj_kernel(
    const float* __restrict__ X, const float* __restrict__ W,
    const float* __restrict__ bias, __bf16* __restrict__ Y, int mode)
{
  __shared__ __align__(16) float xtile[2][64 * 32];
  __shared__ __align__(16) float wtile[2][64 * 32];

  const int lane = threadIdx.x & 31;
  const int wave = threadIdx.x >> 5;
  const int row0 = blockIdx.x * 64;          // block M base
  const int col0 = blockIdx.y * 64;          // block N base
  const int mA   = lane & 15;
  const int nl   = lane & 15;
  const int mhi  = (lane & 16) ? 8 : 0;

  const float* Xblk = X + (size_t)row0 * D_;
  const float* Wblk = W + (size_t)col0 * D_;

#if HAVE_TDM
  const unsigned xoff[2] = {(unsigned)(size_t)&xtile[0][0], (unsigned)(size_t)&xtile[1][0]};
  const unsigned woff[2] = {(unsigned)(size_t)&wtile[0][0], (unsigned)(size_t)&wtile[1][0]};
  if (wave == 0) {
    tdm_load_2d(xoff[0], Xblk, 2u, 32, 64, D_, 4096, D_);
    tdm_load_2d(woff[0], Wblk, 2u, 32, 64, D_, D_, D_);
  }
#endif

  v8f acc[4] = {};

  for (int step = 0; step < D_ / 32; ++step) {
    const int cur = step & 1;
    const int k0  = step * 32;
#if HAVE_TDM
    if (wave == 0) tdm_wait();                 // buffer `cur` landed
    __syncthreads();                           // publish to all waves
    if (wave == 0 && k0 + 32 < D_) {           // prefetch next step into cur^1
      tdm_load_2d(xoff[cur ^ 1], Xblk + k0 + 32, 2u, 32, 64, D_, 4096, D_);
      tdm_load_2d(woff[cur ^ 1], Wblk + k0 + 32, 2u, 32, 64, D_, D_, D_);
    }
#else
    {
      // cooperative copy: 64 rows x 32 f32 per tile, 128 threads
      const int r = threadIdx.x >> 1, half = (threadIdx.x & 1) * 16;
      const float4* xs = reinterpret_cast<const float4*>(Xblk + (size_t)r * D_ + k0 + half);
      const float4* ws = reinterpret_cast<const float4*>(Wblk + (size_t)r * D_ + k0 + half);
      float4* xd = reinterpret_cast<float4*>(&xtile[cur][r * 32 + half]);
      float4* wd = reinterpret_cast<float4*>(&wtile[cur][r * 32 + half]);
#pragma unroll
      for (int i = 0; i < 4; ++i) { xd[i] = xs[i]; wd[i] = ws[i]; }
    }
    __syncthreads();
#endif

    v16bf a  = frag_from_f32(&xtile[cur][(wave * 16 + mA) * 32], 0, lane);
    v16bf b0 = frag_from_f32(&wtile[cur][( 0 + mA) * 32], 0, lane);
    v16bf b1 = frag_from_f32(&wtile[cur][(16 + mA) * 32], 0, lane);
    v16bf b2 = frag_from_f32(&wtile[cur][(32 + mA) * 32], 0, lane);
    v16bf b3 = frag_from_f32(&wtile[cur][(48 + mA) * 32], 0, lane);
    acc[0] = WMMA_BF16(a, b0, acc[0]);
    acc[1] = WMMA_BF16(a, b1, acc[1]);
    acc[2] = WMMA_BF16(a, b2, acc[2]);
    acc[3] = WMMA_BF16(a, b3, acc[3]);

    __syncthreads();                           // done reading buffer `cur`
  }

#pragma unroll
  for (int nt = 0; nt < 4; ++nt) {
    const int c  = col0 + nt * 16 + nl;
    const int h  = c >> 6;
    const int hd = c & 63;
    const float bv = bias[c];
#pragma unroll
    for (int v = 0; v < 8; ++v) {
      const int r = row0 + wave * 16 + v + mhi;
      const int b = r >> 10;
      const int t = r & 1023;
      const float val = acc[nt][v] + bv;
      size_t idx;
      if (mode == 0) idx = (((size_t)(b * H_ + h) * TA_) + t) * HD_ + hd;
      else           idx = (((size_t)(b * H_ + h) * HD_) + hd) * TV_ + t;
      Y[idx] = (__bf16)val;
    }
  }
}

// ---------------------------------------------------------------------------
// Kernel 2: flash attention with transport bias.
// grid = (B*H, TA/128); block = 256 (8 waves); wave owns a 16-row Q tile.
// K (32x64) and V^T (64x32) tiles are TDM double-buffered in LDS.
// ---------------------------------------------------------------------------
__global__ __launch_bounds__(256) void attn_kernel(
    const __bf16* __restrict__ Qw, const __bf16* __restrict__ Kw,
    const __bf16* __restrict__ Vt, const float* __restrict__ guide,
    const float* __restrict__ bias_scale, __bf16* __restrict__ Ctx)
{
  __shared__ __align__(16) __bf16 pbuf[8 * 16 * 32];     // per-wave P relayout slabs
  __shared__ __align__(16) __bf16 ktile[2][32 * HD_];    // K rows  [kv][hd]
  __shared__ __align__(16) __bf16 vtile[2][HD_ * 32];    // V^T rows [hd][kv]

  const int lane = threadIdx.x & 31;
  const int wave = threadIdx.x >> 5;
  const int bh   = blockIdx.x;
  const int b    = bh >> 3;
  const int h    = bh & 7;
  const int q0   = blockIdx.y * 128 + wave * 16;
  const int mA   = lane & 15;
  const int nl   = lane & 15;
  const int mhi  = (lane & 16) ? 8 : 0;

  const float bs = bias_scale[0];
  const float sp = (bs > 20.f) ? bs : log1pf(__expf(bs));   // softplus

  const __bf16* Qbase = Qw + (size_t)bh * TA_ * HD_;
  const __bf16* Kbase = Kw + (size_t)bh * TV_ * HD_;
  const __bf16* Vbase = Vt + (size_t)bh * HD_ * TV_;
  const float*  gbase = guide + ((size_t)b * TA_ + q0) * (size_t)TV_;

  // Q operand for the whole head dim (two K=32 steps), kept resident
  const __bf16* qrow = Qbase + (size_t)(q0 + mA) * HD_;
  const v16bf aq0 = frag_from_bf16(qrow, 0, lane);
  const v16bf aq1 = frag_from_bf16(qrow, 32, lane);

  v8f acc[4] = {};
  float mi[8], li[8];
#pragma unroll
  for (int v = 0; v < 8; ++v) { mi[v] = -1e30f; li[v] = 0.f; }

  __bf16* myp = pbuf + wave * (16 * 32);

#if HAVE_TDM
  const unsigned koff[2] = {(unsigned)(size_t)&ktile[0][0], (unsigned)(size_t)&ktile[1][0]};
  const unsigned voff[2] = {(unsigned)(size_t)&vtile[0][0], (unsigned)(size_t)&vtile[1][0]};
  if (wave == 0) {
    tdm_load_2d(koff[0], Kbase, 1u, HD_, 32, HD_, TV_, HD_);
    tdm_load_2d(voff[0], Vbase, 1u, 32, HD_, TV_, HD_, TV_);
  }
#endif

  for (int kv0 = 0; kv0 < TV_; kv0 += 32) {
    const int cur = (kv0 >> 5) & 1;
#if HAVE_TDM
    if (wave == 0) tdm_wait();                 // tiles `cur` landed
    __syncthreads();                           // publish to all waves
    if (wave == 0 && kv0 + 32 < TV_) {         // prefetch next kv step
      tdm_load_2d(koff[cur ^ 1], Kbase + (size_t)(kv0 + 32) * HD_, 1u, HD_, 32, HD_, TV_, HD_);
      tdm_load_2d(voff[cur ^ 1], Vbase + kv0 + 32, 1u, 32, HD_, TV_, HD_, TV_);
    }
#else
    {
      const int tid = threadIdx.x;
      reinterpret_cast<uint4*>(&ktile[cur][0])[tid] =
          reinterpret_cast<const uint4*>(Kbase + (size_t)kv0 * HD_)[tid];
      const int vr = tid >> 2, vc = (tid & 3) * 8;
      *reinterpret_cast<uint4*>(&vtile[cur][vr * 32 + vc]) =
          *reinterpret_cast<const uint4*>(Vbase + (size_t)vr * TV_ + kv0 + vc);
    }
    __syncthreads();
#endif

    // prefetch next guide tile (gfx1250 global_prefetch path)
    if (kv0 + 32 < TV_)
      __builtin_prefetch(gbase + (size_t)mA * TV_ + kv0 + 32 + nl, 0, 1);

    // ---- scores: S = (Q K^T) for keys [kv0, kv0+32), operands from LDS ----
    v8f s[2];
#pragma unroll
    for (int nt = 0; nt < 2; ++nt) {
      const __bf16* krow = &ktile[cur][(nt * 16 + mA) * HD_];
      v16bf bk0 = frag_from_bf16(krow, 0, lane);
      v16bf bk1 = frag_from_bf16(krow, 32, lane);
      v8f z = {};
      z = WMMA_BF16(aq0, bk0, z);
      z = WMMA_BF16(aq1, bk1, z);
      s[nt] = z;
    }

    // ---- scale + transport bias ------------------------------------------
#pragma unroll
    for (int nt = 0; nt < 2; ++nt) {
      const int key = kv0 + nt * 16 + nl;
#pragma unroll
      for (int v = 0; v < 8; ++v) {
        const int m = v + mhi;
        const float g = gbase[(size_t)m * TV_ + key];
        s[nt][v] = s[nt][v] * SCALE_ + __logf(g + 1e-8f) * sp;
      }
    }

    // ---- online softmax (row = VGPR index + 8*(lane>=16)) ----------------
    float pr[2][8];
#pragma unroll
    for (int v = 0; v < 8; ++v) {
      float mv = fmaxf(s[0][v], s[1][v]);
#pragma unroll
      for (int off = 1; off < 16; off <<= 1)
        mv = fmaxf(mv, __shfl_xor(mv, off, 16));
      const float nm    = fmaxf(mi[v], mv);
      const float alpha = __expf(mi[v] - nm);
      mi[v] = nm;
      const float p0 = __expf(s[0][v] - nm);
      const float p1 = __expf(s[1][v] - nm);
      pr[0][v] = p0; pr[1][v] = p1;
      float rs = p0 + p1;
#pragma unroll
      for (int off = 1; off < 16; off <<= 1)
        rs += __shfl_xor(rs, off, 16);
      li[v] = li[v] * alpha + rs;
#pragma unroll
      for (int nt = 0; nt < 4; ++nt) acc[nt][v] *= alpha;
    }

    // ---- P: C-layout -> A-layout via per-wave private LDS slab -----------
#pragma unroll
    for (int nt = 0; nt < 2; ++nt)
#pragma unroll
      for (int v = 0; v < 8; ++v)
        myp[(v + mhi) * 32 + nt * 16 + nl] = (__bf16)pr[nt][v];

    v16bf ap = frag_from_bf16(myp + mA * 32, 0, lane);

    // ---- context accumulation: acc += P @ V (V^T rows from LDS) ----------
#pragma unroll
    for (int nt = 0; nt < 4; ++nt) {
      const __bf16* vrow = &vtile[cur][(nt * 16 + nl) * 32];
      v16bf bv = frag_from_bf16(vrow, 0, lane);
      acc[nt] = WMMA_BF16(ap, bv, acc[nt]);
    }

    __syncthreads();                           // done reading tiles `cur`
  }

  // ---- normalize and store context [B,TA,D] bf16 -------------------------
#pragma unroll
  for (int nt = 0; nt < 4; ++nt) {
    const int c = h * HD_ + nt * 16 + nl;
#pragma unroll
    for (int v = 0; v < 8; ++v) {
      const int m = v + mhi;
      const int t = q0 + m;
      Ctx[((size_t)b * TA_ + t) * D_ + c] = (__bf16)(acc[nt][v] / li[v]);
    }
  }
}

// ---------------------------------------------------------------------------
// Kernel 3: out = LN(q + sigmoid(gate) * (Ctx @ Wo^T + bo))
// grid = 256 blocks; block = 256 threads (8 waves) owns 16 full rows (D=512).
// ---------------------------------------------------------------------------
__global__ __launch_bounds__(256) void outproj_ln_kernel(
    const __bf16* __restrict__ Ctx, const float* __restrict__ Wo,
    const float* __restrict__ bo, const float* __restrict__ qin,
    const float* __restrict__ gamma, const float* __restrict__ beta,
    const float* __restrict__ gate, float* __restrict__ out)
{
  __shared__ __align__(16) float zbuf[16 * D_];
  __shared__ float rsum[16][17];
  __shared__ float rsq[16][17];
  __shared__ float mu[16];
  __shared__ float rstd[16];

  const int lane = threadIdx.x & 31;
  const int wave = threadIdx.x >> 5;
  const int r0   = blockIdx.x * 16;
  const int col0 = wave * 64;
  const int mA   = lane & 15;
  const int nl   = lane & 15;
  const int mhi  = (lane & 16) ? 8 : 0;

  const float sg = 1.f / (1.f + __expf(-gate[0]));

  v8f acc[4] = {};
  const __bf16* arow = Ctx + (size_t)(r0 + mA) * D_;
  const float* brow0 = Wo + (size_t)(col0 +  0 + mA) * D_;
  const float* brow1 = Wo + (size_t)(col0 + 16 + mA) * D_;
  const float* brow2 = Wo + (size_t)(col0 + 32 + mA) * D_;
  const float* brow3 = Wo + (size_t)(col0 + 48 + mA) * D_;

  for (int k0 = 0; k0 < D_; k0 += 32) {
    v16bf a  = frag_from_bf16(arow, k0, lane);
    v16bf b0 = frag_from_f32(brow0, k0, lane);
    v16bf b1 = frag_from_f32(brow1, k0, lane);
    v16bf b2 = frag_from_f32(brow2, k0, lane);
    v16bf b3 = frag_from_f32(brow3, k0, lane);
    acc[0] = WMMA_BF16(a, b0, acc[0]);
    acc[1] = WMMA_BF16(a, b1, acc[1]);
    acc[2] = WMMA_BF16(a, b2, acc[2]);
    acc[3] = WMMA_BF16(a, b3, acc[3]);
  }

  // gated residual into LDS row buffer
#pragma unroll
  for (int nt = 0; nt < 4; ++nt) {
    const int c = col0 + nt * 16 + nl;
    const float bias = bo[c];
#pragma unroll
    for (int v = 0; v < 8; ++v) {
      const int m = v + mhi;
      const float y = acc[nt][v] + bias;
      const float z = qin[(size_t)(r0 + m) * D_ + c] + sg * y;
      zbuf[m * D_ + c] = z;
    }
  }
  __syncthreads();

  // LayerNorm: 16 threads per row, each covers 32 columns
  const int r   = threadIdx.x & 15;
  const int sgp = threadIdx.x >> 4;   // 0..15
  float s1 = 0.f, s2 = 0.f;
  for (int c = sgp * 32; c < sgp * 32 + 32; ++c) {
    const float z = zbuf[r * D_ + c];
    s1 += z; s2 += z * z;
  }
  rsum[r][sgp] = s1; rsq[r][sgp] = s2;
  __syncthreads();
  if (sgp == 0) {
    float t1 = 0.f, t2 = 0.f;
#pragma unroll
    for (int i = 0; i < 16; ++i) { t1 += rsum[r][i]; t2 += rsq[r][i]; }
    const float m_  = t1 / (float)D_;
    const float var = t2 / (float)D_ - m_ * m_;
    mu[r]   = m_;
    rstd[r] = rsqrtf(var + LN_EPS_);
  }
  __syncthreads();
  const float m_ = mu[r];
  const float rs = rstd[r];
  for (int c = sgp * 32; c < sgp * 32 + 32; ++c) {
    const float z = zbuf[r * D_ + c];
    out[(size_t)(r0 + r) * D_ + c] = (z - m_) * rs * gamma[c] + beta[c];
  }
}

// ---------------------------------------------------------------------------
extern "C" void kernel_launch(void* const* d_in, const int* in_sizes, int n_in,
                              void* d_out, int out_size, void* d_ws, size_t ws_size,
                              hipStream_t stream) {
  const float* q      = (const float*)d_in[0];
  const float* k      = (const float*)d_in[1];
  const float* v      = (const float*)d_in[2];
  const float* guide  = (const float*)d_in[3];
  const float* Wq     = (const float*)d_in[4];
  const float* bq     = (const float*)d_in[5];
  const float* Wk     = (const float*)d_in[6];
  const float* bk     = (const float*)d_in[7];
  const float* Wv     = (const float*)d_in[8];
  const float* bv     = (const float*)d_in[9];
  const float* Wo     = (const float*)d_in[10];
  const float* bo     = (const float*)d_in[11];
  const float* gamma  = (const float*)d_in[12];
  const float* beta   = (const float*)d_in[13];
  const float* gate   = (const float*)d_in[14];
  const float* bscale = (const float*)d_in[15];
  float* out = (float*)d_out;

  // workspace: Qw | Kw | Vt | Ctx, each B*H*T*HD = 2M bf16 elements (4 MB)
  const size_t elems = (size_t)B_ * H_ * TA_ * HD_;
  __bf16* Qw  = (__bf16*)d_ws;
  __bf16* Kw  = Qw + elems;
  __bf16* Vt  = Kw + elems;
  __bf16* Ctx = Vt + elems;

  dim3 pgrid(64, 8), pblock(128);
  proj_kernel<<<pgrid, pblock, 0, stream>>>(q, Wq, bq, Qw, 0);
  proj_kernel<<<pgrid, pblock, 0, stream>>>(k, Wk, bk, Kw, 0);
  proj_kernel<<<pgrid, pblock, 0, stream>>>(v, Wv, bv, Vt, 1);

  attn_kernel<<<dim3(B_ * H_, TA_ / 128), dim3(256), 0, stream>>>(
      Qw, Kw, Vt, guide, bscale, Ctx);

  outproj_ln_kernel<<<dim3((B_ * TA_) / 16), dim3(256), 0, stream>>>(
      Ctx, Wo, bo, q, gamma, beta, gate, out);
}
